// CausalSelfAttention_25417616458349
// MI455X (gfx1250) — matrix operand order repaired and checked by reading, not compiled, not run on hardware
//
#include <hip/hip_runtime.h>
#include <hip/hip_bf16.h>

// ---------------------------------------------------------------------------
// CausalSelfAttention (GQA + RoPE) for gfx1250 (MI455X).
// bf16 WMMA pipeline + Tensor Data Mover staging + prefetch.
// B=2, T=2048, C=2048, H=16, Hkv=4, hd=128.
// ---------------------------------------------------------------------------

typedef __attribute__((ext_vector_type(16))) __bf16 v16bf;
typedef __attribute__((ext_vector_type(8)))  __bf16 v8bf;
typedef __attribute__((ext_vector_type(8)))  float  v8f;
typedef __attribute__((ext_vector_type(4)))  unsigned int v4u;
typedef __attribute__((ext_vector_type(8)))  int v8i;
typedef __attribute__((ext_vector_type(4)))  int v4i;

union Frag16 { v16bf v; v8bf h[2]; };

__device__ __forceinline__ unsigned short f2bf(float f) {
  unsigned u = __builtin_bit_cast(unsigned, f);
  u += 0x7FFFu + ((u >> 16) & 1u);   // round-to-nearest-even
  return (unsigned short)(u >> 16);
}
__device__ __forceinline__ float bf2f(unsigned short h) {
  unsigned u = ((unsigned)h) << 16;
  return __builtin_bit_cast(float, u);
}

// ------------------------- Tensor Data Mover (TDM) ---------------------------
#if defined(__AMDGCN__) && __has_builtin(__builtin_amdgcn_tensor_load_to_lds) && \
    __has_builtin(__builtin_amdgcn_s_wait_tensorcnt)
#define USE_TDM 1
#else
#define USE_TDM 0
#endif

#if USE_TDM
// Issue a 2D tile load (global -> LDS, packed rows) through the TDM.
// tile_w elems (2B each) contiguous per row, tile_h rows, row stride in elems.
// D# packing per CDNA5 ISA ch.8 (group0: 128b, group1: 256b).
__device__ __forceinline__ void tdm_load_2d_bf16(unsigned lds_off,
                                                 const void* gptr,
                                                 unsigned tile_w,
                                                 unsigned tile_h,
                                                 unsigned row_stride) {
  const unsigned long long ga = (unsigned long long)gptr;
  v4u g0;
  g0[0] = 1u;                                   // count=1, is_restore=0
  g0[1] = lds_off;                              // lds_addr (bytes)
  g0[2] = (unsigned)(ga & 0xFFFFFFFFu);         // global_addr[31:0]
  g0[3] = (unsigned)((ga >> 32) & 0x1FFFFFFu)   // global_addr[56:32]
          | (2u << 30);                         // type = 2 ("image")
  v8i g1;
  g1[0] = (int)(1u << 16);                      // data_size = 1 -> 2 bytes
  g1[1] = (int)((tile_w & 0xFFFFu) << 16);      // tensor_dim0[15:0]
  g1[2] = (int)((tile_w >> 16) | ((tile_h & 0xFFFFu) << 16)); // d0 hi | d1 lo
  g1[3] = (int)((tile_h >> 16) | ((tile_w & 0xFFFFu) << 16)); // d1 hi | tile_dim0
  g1[4] = (int)(tile_h & 0xFFFFu);              // tile_dim1 | tile_dim2=0
  g1[5] = (int)row_stride;                      // tensor_dim0_stride[31:0]
  g1[6] = 0;                                    // stride hi | dim1_stride lo
  g1[7] = 0;
  const v4i gz = {0, 0, 0, 0};
#if __clang_major__ >= 23
  const v8i gz8 = {0, 0, 0, 0, 0, 0, 0, 0};
  __builtin_amdgcn_tensor_load_to_lds(g0, g1, gz, gz, gz8, 0);
#else
  __builtin_amdgcn_tensor_load_to_lds(g0, g1, gz, gz, 0);
#endif
}
__device__ __forceinline__ unsigned lds_offset(const void* p) {
  // generic pointer to LDS: low 32 bits are the LDS byte offset (aperture map)
  return (unsigned)(unsigned long long)p;
}
#endif

// ------------------------------- fp32 -> bf16 -------------------------------
__global__ __launch_bounds__(256) void cvt_bf16_kernel(
    const float* __restrict__ in, unsigned short* __restrict__ out, int n4) {
  const int i = blockIdx.x * blockDim.x + threadIdx.x;
  if (i < n4) {
    const float4 f = ((const float4*)in)[i];
    union { unsigned long long u; unsigned short s[4]; } o;
    o.s[0] = f2bf(f.x); o.s[1] = f2bf(f.y);
    o.s[2] = f2bf(f.z); o.s[3] = f2bf(f.w);
    ((unsigned long long*)out)[i] = o.u;
  }
}

// ------------------------------- GEMM (bf16) --------------------------------
// C[M,N] = A[M,K] (bf16, row-major) @ B[K,N] (bf16, row-major)
// 256 threads (8 waves), 128x128 block tile, K step 32.
// A tile staged by TDM (packed [128][32]); B tile staged transposed manually.
template <bool OUT_F32>
__global__ __launch_bounds__(256) void gemm_bf16_kernel(
    const unsigned short* __restrict__ A, const unsigned short* __restrict__ Bm,
    void* __restrict__ Cp, int M, int N, int K) {
  __shared__ __align__(16) unsigned short As[128][32];  // [m][k] packed (TDM)
  __shared__ __align__(16) unsigned short Bs[128][40];  // transposed: [n][k]
  const int tid  = threadIdx.x;
  const int lane = tid & 31;
  const int wave = tid >> 5;
  const int wm = wave >> 2;        // 0..1
  const int wn = wave & 3;         // 0..3
  const int bm = blockIdx.y * 128;
  const int bn = blockIdx.x * 128;
  const int lo = lane & 15;
  const int hi = lane >> 4;
  const v8f vzero = {0.f, 0.f, 0.f, 0.f, 0.f, 0.f, 0.f, 0.f};

  v8f acc[4][2];
#pragma unroll
  for (int mt = 0; mt < 4; ++mt)
#pragma unroll
    for (int nt = 0; nt < 2; ++nt) acc[mt][nt] = vzero;

  const int arow = tid >> 2, aseg = tid & 3;     // fallback A staging mapping
  const int brow = tid >> 4, bseg = tid & 15;    // B: 16 rows/round, 16x8 elems
  (void)arow; (void)aseg;

  for (int k0 = 0; k0 < K; k0 += 32) {
#if USE_TDM
    if (wave == 0)
      tdm_load_2d_bf16(lds_offset(&As[0][0]), A + (size_t)bm * K + k0,
                       /*tile_w=*/32, /*tile_h=*/128, /*stride=*/(unsigned)K);
#else
#pragma unroll
    for (int rnd = 0; rnd < 2; ++rnd) {
      const int r = arow + rnd * 64;
      const uint4 d = *(const uint4*)(A + (size_t)(bm + r) * K + k0 + aseg * 8);
      *(uint4*)(&As[r][aseg * 8]) = d;
    }
#endif
#pragma unroll
    for (int rnd = 0; rnd < 2; ++rnd) {
      const int kk = brow + rnd * 16;
      const uint4 d = *(const uint4*)(Bm + (size_t)(k0 + kk) * N + bn + bseg * 8);
      const unsigned short* ds = (const unsigned short*)&d;
#pragma unroll
      for (int i = 0; i < 8; ++i) Bs[bseg * 8 + i][kk] = ds[i];
    }
    // prefetch next K-step's B tile rows into cache (global_prefetch_b8)
    if (k0 + 32 < K)
      __builtin_prefetch(Bm + (size_t)(k0 + 32 + brow) * N + bn + bseg * 8, 0, 1);
#if USE_TDM
    if (wave == 0) __builtin_amdgcn_s_wait_tensorcnt(0);
#endif
    __syncthreads();

    // A fragment (16x32 bf16, ISA 7.12.2): lane row = lo, K-half by lane>>4.
    Frag16 af[4], bf[2];
    const int basek = hi * 8;
#pragma unroll
    for (int mt = 0; mt < 4; ++mt) {
      const unsigned short* p = &As[wm * 64 + mt * 16 + lo][0];
      af[mt].h[0] = *(const v8bf*)(p + basek);
      af[mt].h[1] = *(const v8bf*)(p + basek + 16);
    }
  // B fragment (32x16 bf16): lane col = lo, k = hi*8..+7 and 16+hi*8..+7.
  const int kb = hi * 8;
#pragma unroll
    for (int nt = 0; nt < 2; ++nt) {
      const unsigned short* p = &Bs[wn * 32 + nt * 16 + lo][0];
      bf[nt].h[0] = *(const v8bf*)(p + kb);
      bf[nt].h[1] = *(const v8bf*)(p + kb + 16);
    }
#pragma unroll
    for (int mt = 0; mt < 4; ++mt)
#pragma unroll
      for (int nt = 0; nt < 2; ++nt)
        acc[mt][nt] = __builtin_amdgcn_wmma_f32_16x16x32_bf16(
            false, af[mt].v, false, bf[nt].v, (short)0, acc[mt][nt], false, false);
    __syncthreads();
  }

#pragma unroll
  for (int mt = 0; mt < 4; ++mt)
#pragma unroll
    for (int nt = 0; nt < 2; ++nt)
#pragma unroll
      for (int r = 0; r < 8; ++r) {
        const int row = bm + wm * 64 + mt * 16 + hi * 8 + r;
        const int col = bn + wn * 32 + nt * 16 + lo;
        const float val = acc[mt][nt][r];
        if (OUT_F32) ((float*)Cp)[(size_t)row * N + col] = val;
        else ((unsigned short*)Cp)[(size_t)row * N + col] = f2bf(val);
      }
}

// --------------------------------- RoPE -------------------------------------
__global__ __launch_bounds__(256) void rope_kernel(
    unsigned short* __restrict__ Q, unsigned short* __restrict__ Kb) {
  const int i = blockIdx.x * blockDim.x + threadIdx.x;
  if (i >= 4096 * 1280) return;            // (B*T) * (16+4 heads) * 64 pairs
  const int bt = i / 1280;
  const int rem = i - bt * 1280;
  const int head = rem >> 6;
  const int p = rem & 63;
  const int t = bt & 2047;
  const float inv = __expf(-(float)p * 0.14391156531310085f); // 10000^(-p/64)
  const float ang = (float)t * inv;
  float sn, cs;
  __sincosf(ang, &sn, &cs);
  unsigned short* ptr;
  if (head < 16) ptr = Q + (size_t)bt * 2048 + head * 128;
  else           ptr = Kb + (size_t)bt * 512 + (head - 16) * 128;
  const float x1 = bf2f(ptr[p]);
  const float x2 = bf2f(ptr[p + 64]);
  ptr[p]      = f2bf(x1 * cs - x2 * sn);
  ptr[p + 64] = f2bf(x2 * cs + x1 * sn);
}

// ---------------------------- Flash attention -------------------------------
// grid (T/64, B*H), block 128 (4 waves). Each wave owns 16 query rows.
// K tile staged by TDM (packed [32][128]); V tile staged transposed manually.
__global__ __launch_bounds__(128) void attn_kernel(
    const unsigned short* __restrict__ Q, const unsigned short* __restrict__ Kb,
    const unsigned short* __restrict__ Vb, unsigned short* __restrict__ O) {
  __shared__ __align__(16) unsigned short Ks[32][128];   // [key][d] packed (TDM)
  __shared__ __align__(16) unsigned short VTs[128][40];  // [d][key]
  __shared__ __align__(16) unsigned short Ps[4][16][40]; // per-wave P tile
  const int tid  = threadIdx.x;
  const int lane = tid & 31;
  const int wave = tid >> 5;
  const int lo = lane & 15, hi = lane >> 4;
  const int bh = blockIdx.y;
  const int b = bh >> 4, h = bh & 15;
  const int hk = h >> 2;                    // GQA: kv head = h / 4
  const int q0 = blockIdx.x * 64;
  const int qr = q0 + wave * 16;
  const float scale = 0.08838834764831845f; // 1/sqrt(128)
  const v8f vzero = {0.f, 0.f, 0.f, 0.f, 0.f, 0.f, 0.f, 0.f};

  // Q fragments: 16 rows x 128, as 4 chained 16x32 A-fragments. Kept in regs.
  Frag16 qf[4];
  const int basek = hi * 8;
  const unsigned short* qp = Q + ((size_t)(b * 2048 + qr + lo)) * 2048 + h * 128;
#pragma unroll
  for (int c = 0; c < 4; ++c) {
    qf[c].h[0] = *(const v8bf*)(qp + c * 32 + basek);
    qf[c].h[1] = *(const v8bf*)(qp + c * 32 + basek + 16);
  }

  v8f o[8];
#pragma unroll
  for (int d = 0; d < 8; ++d) o[d] = vzero;
  float rmax[8], rsum[8];
#pragma unroll
  for (int r = 0; r < 8; ++r) { rmax[r] = -__builtin_inff(); rsum[r] = 0.f; }

  const int srow = tid >> 4, sseg = tid & 15;
  const int jend = q0 + 64;
  for (int j = 0; j < jend; j += 32) {
    // stage K via TDM (rows packed), V transposed manually
#if USE_TDM
    if (wave == 0)
      tdm_load_2d_bf16(lds_offset(&Ks[0][0]),
                       Kb + ((size_t)(b * 2048 + j)) * 512 + hk * 128,
                       /*tile_w=*/128, /*tile_h=*/32, /*stride=*/512);
#endif
#pragma unroll
    for (int rnd = 0; rnd < 4; ++rnd) {
      const int rr = srow + rnd * 8;
      const size_t base = ((size_t)(b * 2048 + j + rr)) * 512 + hk * 128 + sseg * 8;
#if !USE_TDM
      const uint4 dk = *(const uint4*)(Kb + base);
      *(uint4*)(&Ks[rr][sseg * 8]) = dk;
#endif
      const uint4 dv = *(const uint4*)(Vb + base);
      const unsigned short* ds = (const unsigned short*)&dv;
#pragma unroll
      for (int i = 0; i < 8; ++i) VTs[sseg * 8 + i][rr] = ds[i];
    }
#if USE_TDM
    if (wave == 0) __builtin_amdgcn_s_wait_tensorcnt(0);
#endif
    __syncthreads();

    // S = Q @ K^T : two 16x16 tiles, each from 4 chained k=32 WMMAs
    v8f s[2];
#pragma unroll
    for (int nt = 0; nt < 2; ++nt) {
      v8f a = vzero;
#pragma unroll
      for (int c = 0; c < 4; ++c) {
        Frag16 kf;
        const unsigned short* kp = &Ks[nt * 16 + lo][c * 32 + hi * 8];
        kf.h[0] = *(const v8bf*)(kp);
        kf.h[1] = *(const v8bf*)(kp + 16);
        a = __builtin_amdgcn_wmma_f32_16x16x32_bf16(false, qf[c].v, false, kf.v,
                                                    (short)0, a, false, false);
      }
      s[nt] = a;
    }
    // causal mask + scale
#pragma unroll
    for (int nt = 0; nt < 2; ++nt)
#pragma unroll
      for (int r = 0; r < 8; ++r) {
        const int qi = qr + hi * 8 + r;
        const int ki = j + nt * 16 + lo;
        const float v = s[nt][r] * scale;
        s[nt][r] = (ki <= qi) ? v : -__builtin_inff();
      }
    // online softmax (16-lane shuffle reductions; halves hold distinct rows)
    float alpha[8];
#pragma unroll
    for (int r = 0; r < 8; ++r) {
      float m = fmaxf(s[0][r], s[1][r]);
#pragma unroll
      for (int off = 1; off < 16; off <<= 1) m = fmaxf(m, __shfl_xor(m, off, 32));
      const float nm = fmaxf(rmax[r], m);
      alpha[r] = __expf(rmax[r] - nm);
      const float p0 = __expf(s[0][r] - nm);
      const float p1 = __expf(s[1][r] - nm);
      s[0][r] = p0; s[1][r] = p1;
      float ps = p0 + p1;
#pragma unroll
      for (int off = 1; off < 16; off <<= 1) ps += __shfl_xor(ps, off, 32);
      rsum[r] = rsum[r] * alpha[r] + ps;
      rmax[r] = nm;
    }
#pragma unroll
    for (int d = 0; d < 8; ++d)
#pragma unroll
      for (int r = 0; r < 8; ++r) o[d][r] *= alpha[r];

    // Re-layout P (C-layout -> A-fragment) via per-wave LDS tile.
    // LDS ops from one wave complete in order (ISA 7.3), no barrier needed.
#pragma unroll
    for (int nt = 0; nt < 2; ++nt)
#pragma unroll
      for (int r = 0; r < 8; ++r)
        Ps[wave][hi * 8 + r][nt * 16 + lo] = f2bf(s[nt][r]);
    Frag16 pf;
    {
      const unsigned short* pp = &Ps[wave][lo][0];
      pf.h[0] = *(const v8bf*)(pp + basek);
      pf.h[1] = *(const v8bf*)(pp + basek + 16);
    }
    // O += P @ V : 8 WMMAs across head_dim
#pragma unroll
    for (int d = 0; d < 8; ++d) {
      Frag16 vf;
      const unsigned short* vp = &VTs[d * 16 + lo][hi * 8];
      vf.h[0] = *(const v8bf*)(vp);
      vf.h[1] = *(const v8bf*)(vp + 16);
      o[d] = __builtin_amdgcn_wmma_f32_16x16x32_bf16(false, pf.v, false, vf.v,
                                                     (short)0, o[d], false, false);
    }
    __syncthreads();
  }

  // normalize + store bf16 attention output [B,T,C] (head-interleaved)
#pragma unroll
  for (int d = 0; d < 8; ++d)
#pragma unroll
    for (int r = 0; r < 8; ++r) {
      const int m = hi * 8 + r;
      const float val = o[d][r] / rsum[r];
      O[((size_t)(b * 2048 + qr + m)) * 2048 + h * 128 + d * 16 + lo] = f2bf(val);
    }
}

// ------------------------------- launcher -----------------------------------
extern "C" void kernel_launch(void* const* d_in, const int* in_sizes, int n_in,
                              void* d_out, int out_size, void* d_ws, size_t ws_size,
                              hipStream_t stream) {
  (void)in_sizes; (void)n_in; (void)out_size; (void)ws_size;
  const float* x  = (const float*)d_in[0];
  const float* Wq = (const float*)d_in[1];
  const float* Wk = (const float*)d_in[2];
  const float* Wv = (const float*)d_in[3];
  const float* Wo = (const float*)d_in[4];

  char* ws = (char*)d_ws;
  unsigned short* xb  = (unsigned short*)(ws);                    // 16 MB
  unsigned short* wqb = (unsigned short*)(ws + 16777216);         //  8 MB
  unsigned short* wkb = (unsigned short*)(ws + 25165824);         //  2 MB
  unsigned short* wvb = (unsigned short*)(ws + 27262976);         //  2 MB
  unsigned short* wob = (unsigned short*)(ws + 29360128);         //  8 MB
  unsigned short* qb  = (unsigned short*)(ws + 37748736);         // 16 MB
  unsigned short* kb  = (unsigned short*)(ws + 54525952);         //  4 MB
  unsigned short* vb  = (unsigned short*)(ws + 58720256);         //  4 MB
  unsigned short* ob  = (unsigned short*)(ws + 62914560);         // 16 MB

  auto cvt = [&](const float* in, unsigned short* out, int n) {
    const int n4 = n >> 2;
    cvt_bf16_kernel<<<(n4 + 255) / 256, 256, 0, stream>>>(in, out, n4);
  };
  cvt(x,  xb,  8388608);
  cvt(Wq, wqb, 4194304);
  cvt(Wk, wkb, 1048576);
  cvt(Wv, wvb, 1048576);
  cvt(Wo, wob, 4194304);

  // QKV projections
  gemm_bf16_kernel<false><<<dim3(16, 32), dim3(256), 0, stream>>>(xb, wqb, qb, 4096, 2048, 2048);
  gemm_bf16_kernel<false><<<dim3(4, 32),  dim3(256), 0, stream>>>(xb, wkb, kb, 4096, 512, 2048);
  gemm_bf16_kernel<false><<<dim3(4, 32),  dim3(256), 0, stream>>>(xb, wvb, vb, 4096, 512, 2048);

  // RoPE on Q and K
  rope_kernel<<<(4096 * 1280 + 255) / 256, 256, 0, stream>>>(qb, kb);

  // Flash attention
  attn_kernel<<<dim3(32, 32), dim3(128), 0, stream>>>(qb, kb, vb, ob);

  // Output projection (fp32 output)
  gemm_bf16_kernel<true><<<dim3(16, 32), dim3(256), 0, stream>>>(ob, wob, d_out, 4096, 2048, 2048);
}
